// Joint_module_5428838662369
// MI455X (gfx1250) — matrix operand order repaired and checked by reading, not compiled, not run on hardware
//
#include <hip/hip_runtime.h>

// Problem constants (from reference): B=32, C=128, H=W=96
#define HWPIX 9216      // H*W
#define NK    384       // 3*C input channels
#define NC    128       // C
#define NBATCH 32
#define EPSV  1e-5f

typedef float  v8f   __attribute__((ext_vector_type(8)));
typedef __bf16 v16bf __attribute__((ext_vector_type(16)));
typedef unsigned int u32x4 __attribute__((ext_vector_type(4)));
typedef unsigned int u32x8 __attribute__((ext_vector_type(8)));

union Frag { v16bf v; unsigned int d[8]; u32x4 q[2]; };

// one-instruction f32x2 -> packed bf16x2 (truncating): v_perm_b32
__device__ __forceinline__ unsigned int bfpack2(float lo, float hi) {
  return __builtin_amdgcn_perm(__float_as_uint(hi), __float_as_uint(lo), 0x07060302u);
}
__device__ __forceinline__ unsigned short bf1(float x) {   // RNE, for weights (once)
  unsigned int a = __float_as_uint(x);
  return (unsigned short)((a + 0x7FFFu + ((a >> 16) & 1u)) >> 16);
}
__device__ __forceinline__ v8f zero8() {
  v8f z = {0.f,0.f,0.f,0.f,0.f,0.f,0.f,0.f}; return z;
}
__device__ __forceinline__ void loadB(Frag& f, const u32x4* __restrict__ p) {
  f.q[0] = p[0]; f.q[1] = p[2];          // 2x ds_load_b128, 16B aligned
}
// A-frag (16px x 32ch) from global, pixel==lane (coalesced), 8 v_perm packs.
// Two base pointers keep all byte offsets within the 24-bit signed immediate.
__device__ __forceinline__ Frag load_a(const float* __restrict__ p0,
                                       const float* __restrict__ p1, int kb) {
  Frag a;
#pragma unroll
  for (int half = 0; half < 2; ++half)
#pragma unroll
    for (int j = 0; j < 4; ++j) {
      const int k = kb + half * 16 + 2 * j;
      const float* p = (k < 192) ? p0 : p1;
      const int ko = (k < 192) ? k : (k - 192);
      a.d[half * 4 + j] = bfpack2(p[(size_t)ko * HWPIX], p[(size_t)(ko + 1) * HWPIX]);
    }
  return a;
}

// LDS layout (bytes). Weight rows = 392 bf16 (196-dword stride): every frag base is
// 16B aligned (b128 loads) and a 32-lane b128 covers all 64 banks exactly once.
// W1s : [128][392] bf16   Was : [64][392] bf16   Wbs : [128][72] bf16
// Hs  : 8 waves x [16][72] bf16     STG : 32KB fp32 staging for TDM copy of Wb
#define OFF_W1  0
#define OFF_WA  100352
#define OFF_WB  150528
#define OFF_H   168960
#define OFF_STG 187392
#define LDS_BYTES 220160

extern "C" __global__ __launch_bounds__(256)
void joint_main_kernel(const float* __restrict__ fv,
                       const float* __restrict__ W1, const float* __restrict__ b1,
                       const float* __restrict__ g1, const float* __restrict__ be1,
                       const float* __restrict__ m1, const float* __restrict__ v1,
                       const float* __restrict__ Wa, const float* __restrict__ ba,
                       const float* __restrict__ Wb,
                       float* __restrict__ S1g, float* __restrict__ S2g)
{
  extern __shared__ char smem[];
  unsigned short* W1s = (unsigned short*)(smem + OFF_W1);
  unsigned short* Was = (unsigned short*)(smem + OFF_WA);
  unsigned short* Wbs = (unsigned short*)(smem + OFF_WB);
  unsigned short* Hs  = (unsigned short*)(smem + OFF_H);

  const int tid  = threadIdx.x;
  const int lane = tid & 31;
  const int wave = tid >> 5;
  const int lhi  = lane >> 4;
  const int lm   = lane & 15;

  // ---- TDM: DMA Wb (8192 f32 = 32KB) global -> LDS staging (1-D tile D#) ----
  if (wave == 0) {
    unsigned long long ga = (unsigned long long)(const void*)Wb;
    u32x4 g0;
    g0[0] = 1u;                                   // count=1, user mode
    g0[1] = (unsigned)OFF_STG;                    // lds_addr
    g0[2] = (unsigned)(ga & 0xFFFFFFFFu);         // global_addr[31:0]
    g0[3] = (unsigned)((ga >> 32) & 0x01FFFFFFu) | 0x80000000u; // addr[56:32] | type=2
    u32x8 g1v;
    g1v[0] = 0x00020000u;                         // data_size=4B
    g1v[1] = 0x20000000u;                         // tensor_dim0 = 8192 (bits 63:48)
    g1v[2] = 0x00010000u;                         // tensor_dim1 = 1
    g1v[3] = 0x20000000u;                         // tile_dim0 = 8192 (bits 127:112)
    g1v[4] = 0x00000001u;                         // tile_dim1 = 1
    g1v[5] = 0x00002000u;                         // tensor_dim0_stride = 8192
    g1v[6] = 0u; g1v[7] = 0u;
    asm volatile("tensor_load_to_lds %0, %1" :: "s"(g0), "s"(g1v) : "memory");
  }

  // Overlap: convert W1/Wa to bf16 LDS while the TDM runs.
  for (int i = tid; i < NC * NK; i += 256) { int n = i / NK, k = i % NK; W1s[n*392 + k] = bf1(W1[i]); }
  for (int i = tid; i < 64 * NK; i += 256) { int n = i / NK, k = i % NK; Was[n*392 + k] = bf1(Wa[i]); }

  if (wave == 0) __builtin_amdgcn_s_wait_tensorcnt(0);
  __syncthreads();
  { // convert staged Wb fp32 -> bf16
    const float* stg = (const float*)(smem + OFF_STG);
    for (int i = tid; i < NC * 64; i += 256) { int n = i >> 6, k = i & 63; Wbs[n*72 + k] = bf1(stg[i]); }
  }
  __syncthreads();

  const int b    = blockIdx.y;
  const int slab = blockIdx.x;
  const float* fvb = fv + (size_t)b * NK * HWPIX;
  const unsigned int* W1d = (const unsigned int*)W1s;
  const unsigned int* Wad = (const unsigned int*)Was;
  const unsigned int* Wbd = (const unsigned int*)Wbs;
  unsigned short* Hw = Hs + wave * 16 * 72;

  // Hoisted per-channel BN/bias constants (loop-invariant over pixels)
  float sfac[8], shft[8], bav[4];
#pragma unroll
  for (int t = 0; t < 8; ++t) {
    const int n = 16 * t + lm;
    const float s = g1[n] * rsqrtf(v1[n] + EPSV);
    sfac[t] = s;
    shft[t] = be1[n] - m1[n] * s + b1[n] * s;
  }
#pragma unroll
  for (int t = 0; t < 4; ++t) bav[t] = ba[16 * t + lm];

  float rs1[8], rs2[8];
#pragma unroll
  for (int t = 0; t < 8; ++t) { rs1[t] = 0.f; rs2[t] = 0.f; }

#pragma unroll 1
  for (int it = 0; it < 8; ++it) {
    const int px = slab * 1024 + it * 128 + wave * 16 + lm;
    const float* ap  = fvb + px;                 // channels [0,192)
    const float* ap2 = ap + (size_t)192 * HWPIX; // channels [192,384)

    v8f acc1[8]; v8f acc2[4];
#pragma unroll
    for (int t = 0; t < 8; ++t) acc1[t] = zero8();
#pragma unroll
    for (int t = 0; t < 4; ++t) acc2[t] = zero8();

    // ---- K loop, fully unrolled; A double-buffered (global), B double-buffered (LDS)
    Frag aa[2];
    aa[0] = load_a(ap, ap2, lhi * 8);
#pragma unroll
    for (int ks = 0; ks < 12; ++ks) {
      if (ks < 11) aa[(ks + 1) & 1] = load_a(ap, ap2, (ks + 1) * 32 + lhi * 8);
      const Frag& a = aa[ks & 1];
      const u32x4* bW1 = (const u32x4*)(W1d + lm * 196 + ks * 16 + lhi * 4);
      const u32x4* bWa = (const u32x4*)(Wad + lm * 196 + ks * 16 + lhi * 4);
      Frag bb[2];
      loadB(bb[0], bW1);
#pragma unroll
      for (int t = 0; t < 12; ++t) {
        if (t < 11) {
          const u32x4* p = (t < 7) ? (bW1 + (t + 1) * 784)   // 16 rows * 196 dwords / 4
                                   : (bWa + (t - 7) * 784);
          loadB(bb[(t + 1) & 1], p);
        }
        if (t < 8)
          acc1[t] = __builtin_amdgcn_wmma_f32_16x16x32_bf16(
              false, a.v, false, bb[t & 1].v, (short)0, acc1[t], false, false);
        else
          acc2[t - 8] = __builtin_amdgcn_wmma_f32_16x16x32_bf16(
              false, a.v, false, bb[t & 1].v, (short)0, acc2[t - 8], false, false);
      }
    }

    // ---- branch1: BN + ReLU + per-lane spatial accumulation (shuffle deferred)
#pragma unroll
    for (int t = 0; t < 8; ++t) {
      float r = 0.f;
#pragma unroll
      for (int e = 0; e < 8; ++e) r += fmaxf(acc1[t][e] * sfac[t] + shft[t], 0.f);
      rs1[t] += r;
    }

    // ---- branch2a: ReLU(h+ba) -> LDS (C-layout -> A-layout transpose)
#pragma unroll
    for (int t = 0; t < 4; ++t) {
      const int ch = 16 * t + lm;
#pragma unroll
      for (int e = 0; e < 8; ++e)
        Hw[(e + lhi * 8) * 72 + ch] = bf1(fmaxf(acc2[t][e] + bav[t], 0.f));
    }
    asm volatile("s_wait_dscnt 0" ::: "memory");  // staged h visible before re-read

    // ---- branch2b: Wb (K=64 -> N=128), B double-buffered
    v8f acc3[8];
#pragma unroll
    for (int t = 0; t < 8; ++t) acc3[t] = zero8();
#pragma unroll
    for (int ks2 = 0; ks2 < 2; ++ks2) {
      const int k0 = ks2 * 32;
      Frag a2;
      a2.q[0] = *(const u32x4*)(Hw + lm * 72 + k0 + lhi * 8);
      a2.q[1] = *(const u32x4*)(Hw + lm * 72 + k0 + 16 + lhi * 8);
      const u32x4* bW = (const u32x4*)(Wbd + lm * 36 + (k0 >> 1) + lhi * 4);
      Frag bb2[2];
      loadB(bb2[0], bW);
#pragma unroll
      for (int t = 0; t < 8; ++t) {
        if (t < 7) loadB(bb2[(t + 1) & 1], bW + (t + 1) * 144);  // 16 rows * 36 dw / 4
        acc3[t] = __builtin_amdgcn_wmma_f32_16x16x32_bf16(
            false, a2.v, false, bb2[t & 1].v, (short)0, acc3[t], false, false);
      }
    }
#pragma unroll
    for (int t = 0; t < 8; ++t) {
      float r = 0.f;
#pragma unroll
      for (int e = 0; e < 8; ++e) r += acc3[t][e];
      rs2[t] += r;
    }
  }

  // ---- one shuffle + one global atomic per channel, per wave
#pragma unroll
  for (int t = 0; t < 8; ++t) {
    const float r1 = rs1[t] + __shfl_xor(rs1[t], 16);
    const float r2 = rs2[t] + __shfl_xor(rs2[t], 16);
    if (lane < 16) {
      atomicAdd(&S1g[b * NC + 16 * t + lm], r1);
      atomicAdd(&S2g[b * NC + 16 * t + lm], r2);
    }
  }
}

// Tiny [32,256] epilogue on the spatial means (round-0 algebra: the mean commutes
// with the affine conv1x1+BN, so only mean(fvt) is ever needed).
extern "C" __global__ __launch_bounds__(256)
void joint_epilogue_kernel(const float* __restrict__ ft,
                           const float* __restrict__ Wfc1, const float* __restrict__ bfc1,
                           const float* __restrict__ Wfc2, const float* __restrict__ bfc2,
                           const float* __restrict__ bb,
                           const float* __restrict__ Wj, const float* __restrict__ bj,
                           const float* __restrict__ gj, const float* __restrict__ bej,
                           const float* __restrict__ mj, const float* __restrict__ vj,
                           const float* __restrict__ S1g, const float* __restrict__ S2g,
                           float* __restrict__ out)
{
  __shared__ float U[NBATCH][257];
  const int ch = threadIdx.x;
  const float inv = 1.f / (float)HWPIX;
  for (int b = 0; b < NBATCH; ++b) {
    float u;
    if (ch < NC) {
      float f = bfc1[ch];
      for (int c = 0; c < NC; ++c) f += Wfc1[ch * NC + c] * ft[b * NC + c];
      u = (1.f + f) * (S1g[b * NC + ch] * inv);
    } else {
      const int n = ch - NC;
      float f = bfc2[n];
      for (int c = 0; c < NC; ++c) f += Wfc2[n * NC + c] * ft[b * NC + c];
      const float gate = 1.f / (1.f + __expf(-(S2g[b * NC + n] * inv + bb[n])));
      u = f * (1.f + gate);
    }
    U[b][ch] = u;
  }
  __syncthreads();
  const float s = gj[ch] * rsqrtf(vj[ch] + EPSV);
  const float t = bej[ch] - mj[ch] * s;
  for (int b = 0; b < NBATCH; ++b) {
    float acc = bj[ch];
    for (int c = 0; c < 256; ++c) acc += Wj[ch * 256 + c] * U[b][c];
    out[b * 256 + ch] = U[b][ch] + acc * s + t;
  }
}

extern "C" void kernel_launch(void* const* d_in, const int* in_sizes, int n_in,
                              void* d_out, int out_size, void* d_ws, size_t ws_size,
                              hipStream_t stream) {
  (void)in_sizes; (void)n_in; (void)out_size; (void)ws_size;
  const float* fv   = (const float*)d_in[0];
  const float* ft   = (const float*)d_in[1];
  const float* W1   = (const float*)d_in[2];
  const float* b1   = (const float*)d_in[3];
  const float* g1   = (const float*)d_in[4];
  const float* be1  = (const float*)d_in[5];
  const float* m1   = (const float*)d_in[6];
  const float* v1   = (const float*)d_in[7];
  const float* Wfc1 = (const float*)d_in[8];
  const float* bfc1 = (const float*)d_in[9];
  const float* Wa   = (const float*)d_in[10];
  const float* ba   = (const float*)d_in[11];
  const float* Wb   = (const float*)d_in[12];
  const float* bb   = (const float*)d_in[13];
  const float* Wfc2 = (const float*)d_in[14];
  const float* bfc2 = (const float*)d_in[15];
  const float* Wj   = (const float*)d_in[16];
  const float* bj   = (const float*)d_in[17];
  const float* gj   = (const float*)d_in[18];
  const float* bej  = (const float*)d_in[19];
  const float* mj   = (const float*)d_in[20];
  const float* vj   = (const float*)d_in[21];

  float* S1g = (float*)d_ws;                 // [32][128]
  float* S2g = S1g + NBATCH * NC;            // [32][128]
  hipMemsetAsync(d_ws, 0, 2 * NBATCH * NC * sizeof(float), stream);

  hipFuncSetAttribute(reinterpret_cast<const void*>(joint_main_kernel),
                      hipFuncAttributeMaxDynamicSharedMemorySize, LDS_BYTES);
  joint_main_kernel<<<dim3(9, NBATCH), 256, LDS_BYTES, stream>>>(
      fv, W1, b1, g1, be1, m1, v1, Wa, ba, Wb, S1g, S2g);
  joint_epilogue_kernel<<<1, 256, 0, stream>>>(
      ft, Wfc1, bfc1, Wfc2, bfc2, bb, Wj, bj, gj, bej, mj, vj, S1g, S2g, (float*)d_out);
}
